// E2ECompressedGEBDModel_78151224918744
// MI455X (gfx1250) — matrix operand order repaired and checked by reading, not compiled, not run on hardware
//
#include <hip/hip_runtime.h>
#include <hip/hip_bf16.h>
#include <cstdint>

// MI455X / gfx1250 implementation.
// Dominant cost: 3x (256ch 5x5 conv on 17x17 x 1024 imgs) ~= 2.9 TFLOP -> WMMA f16.
// Conv layers: one workgroup per image; image staged to LDS (async-to-LDS path
// when available), weights streamed once per wave with 19 persistent
// accumulators -> near-minimal L2 traffic. LSTM GEMMs on WMMA as before.

typedef __attribute__((ext_vector_type(16))) _Float16 v16h;
typedef __attribute__((ext_vector_type(8)))  _Float16 v8h;
typedef __attribute__((ext_vector_type(8)))  float    v8f;
typedef __attribute__((ext_vector_type(4)))  int      v4i;

#define KWIN 8
#define DIM 256
#define GROUP 4
#define BB 4
#define TT 256
#define NW 32                   // TT / KWIN
#define NSEQ 1024               // KWIN * BB * NW
#define LW 17                   // 2K+1
#define H 256
#define FOURH 1024
#define NPIX 289                // 17*17
#define MT_TILES 19             // ceil(289/16)
#define IMG_BYTES (NPIX * 256 * 2)  // 147968 B per image, f16 NHWC

#if defined(__has_builtin)
#if __has_builtin(__builtin_amdgcn_global_load_async_to_lds_b128)
#define HAS_ASYNC_LDS 1
#endif
#endif
#ifndef HAS_ASYNC_LDS
#define HAS_ASYNC_LDS 0
#endif

#if HAS_ASYNC_LDS
typedef __attribute__((address_space(1))) v4i* as1_v4i_ptr;
typedef __attribute__((address_space(3))) v4i* as3_v4i_ptr;
#endif

// ---------------------------------------------------------------------------
// WMMA fragment helpers (per CDNA5 ISA layouts, wave32)
// ---------------------------------------------------------------------------
__device__ __forceinline__ v16h frag_cat(v8h a, v8h b) {
  v16h v;
#pragma unroll
  for (int i = 0; i < 8; ++i) { v[i] = a[i]; v[i + 8] = b[i]; }
  return v;
}

__device__ __forceinline__ v16h frag_zero() {
  v16h v;
#pragma unroll
  for (int i = 0; i < 16; ++i) v[i] = (_Float16)0.f;
  return v;
}

// A matrix 16x32 (MxK), row-major source with leading dim ld.
// lane<16: M=lane,    K = k0+[0..7]  and k0+[16..23]
// lane>=16: M=lane-16, K = k0+[8..15] and k0+[24..31]
__device__ __forceinline__ v16h load_fragA(const _Float16* __restrict__ base,
                                           int ld, int row, int k0) {
  int lane = threadIdx.x & 31;
  const _Float16* p =
      base + (size_t)(row + (lane & 15)) * ld + (k0 + ((lane & 16) ? 8 : 0));
  return frag_cat(*(const v8h*)p, *(const v8h*)(p + 16));
}

// B matrix 32x16 (KxN). Source stored as (N,K) row-major (i.e. the weight
// matrix W where D = A*W^T). lane<16: N=lane, K=k0+[0..7] and k0+[16..23];
// lane>=16: N=lane-16, K=k0+[8..15] and k0+[24..31] (same K order as A).
__device__ __forceinline__ v16h load_fragB(const _Float16* __restrict__ base,
                                           int ld, int col, int k0) {
  int lane = threadIdx.x & 31;
  const _Float16* p =
      base + (size_t)(col + (lane & 15)) * ld + (k0 + ((lane & 16) ? 8 : 0));
  return frag_cat(*(const v8h*)p, *(const v8h*)(p + 16));
}

// ---------------------------------------------------------------------------
// Generic WMMA GEMM: D(MxN) = A(MxK) * W(NxK)^T [+ bias0+bias1] [+ Cinit]
// grid = (N/64, M/32), block = 256 (8 waves, 2 M-subtiles x 4 N-subtiles)
// ---------------------------------------------------------------------------
__global__ void wmma_gemm_nk(const _Float16* __restrict__ A, int lda,
                             const _Float16* __restrict__ W, int ldw,
                             const float* __restrict__ bias0,
                             const float* __restrict__ bias1,
                             const float* __restrict__ Cinit, long cstride,
                             float* __restrict__ D, int ldd, int Kdim) {
  int wid = threadIdx.x >> 5;
  int lane = threadIdx.x & 31;
  int row0 = blockIdx.y * 32 + (wid >> 2) * 16;
  int col0 = blockIdx.x * 64 + (wid & 3) * 16;
  v8f acc;
#pragma unroll
  for (int i = 0; i < 8; ++i) acc[i] = 0.f;
  for (int k0 = 0; k0 < Kdim; k0 += 32) {
    v16h a = load_fragA(A, lda, row0, k0);
    v16h b = load_fragB(W, ldw, col0, k0);
    acc = __builtin_amdgcn_wmma_f32_16x16x32_f16(false, a, false, b,
                                                 (short)0, acc, false, false);
  }
  int ncol = col0 + (lane & 15);
  int mbase = row0 + ((lane & 16) ? 8 : 0);
  float badd = 0.f;
  if (bias0) badd += bias0[ncol];
  if (bias1) badd += bias1[ncol];
#pragma unroll
  for (int i = 0; i < 8; ++i) {
    float v = acc[i] + badd;
    if (Cinit) v += Cinit[(size_t)(mbase + i) * cstride + ncol];
    D[(size_t)(mbase + i) * ldd + ncol] = v;
  }
}

// ---------------------------------------------------------------------------
// WMMA conv 5x5 SAME on 17x17, 256ch -> 256ch, NHWC f16 activations.
// One workgroup per image (512 thr = 16 waves, one 16-oc N-tile per wave).
// Image staged once into LDS (async-to-LDS when available). K-loop outermost:
// each B fragment loaded once from global, reused by 19 persistent M-tile
// accumulators whose A fragments come from LDS (ds_load_b128).
// ---------------------------------------------------------------------------
__global__ __launch_bounds__(512) void wmma_conv5x5(
    const _Float16* __restrict__ actin, const _Float16* __restrict__ wpk,
    const float* __restrict__ bns, const float* __restrict__ bnb,
    _Float16* __restrict__ actout) {
  extern __shared__ char smem[];
  int tid = threadIdx.x;
  int wid = tid >> 5;
  int lane = tid & 31;
  size_t imgoff = (size_t)blockIdx.x * NPIX * 256;
  const _Float16* img = actin + imgoff;

  // ---- stage image (147,968 B) into LDS ----
#if HAS_ASYNC_LDS
  for (int idx = tid; idx < IMG_BYTES / 16; idx += 512) {
    __builtin_amdgcn_global_load_async_to_lds_b128(
        (as1_v4i_ptr)(uintptr_t)(img + (size_t)idx * 8),
        (as3_v4i_ptr)(uintptr_t)(smem + idx * 16), 0, 0);
  }
  asm volatile("s_wait_asynccnt 0" ::: "memory");
#else
  for (int idx = tid; idx < IMG_BYTES / 16; idx += 512)
    *(v8h*)(smem + idx * 16) = *(const v8h*)(img + (size_t)idx * 8);
#endif
  __syncthreads();

  int col0 = wid * 16;
  int oc = col0 + (lane & 15);
  int kbhalf = (lane & 16) ? 8 : 0;  // A-frag K-half offset (elements)
  int prow = lane & 15;              // A-frag M row within tile

  v8f acc[MT_TILES];
#pragma unroll
  for (int mt = 0; mt < MT_TILES; ++mt)
#pragma unroll
    for (int i = 0; i < 8; ++i) acc[mt][i] = 0.f;

#pragma unroll 1
  for (int tap = 0; tap < 25; ++tap) {
    int ky = tap / 5 - 2, kx = tap % 5 - 2;
    // per-M-tile LDS byte offsets for this tap (-1 => zero / halo)
    int ro[MT_TILES];
#pragma unroll
    for (int mt = 0; mt < MT_TILES; ++mt) {
      int p = mt * 16 + prow;
      int y = p / 17, x = p % 17;
      int iy = y + ky, ix = x + kx;
      bool ok = (p < NPIX) && ((unsigned)iy < 17u) && ((unsigned)ix < 17u);
      ro[mt] = ok ? ((iy * 17 + ix) * 256 + kbhalf) * 2 : -1;
    }
    const _Float16* wt = wpk + (size_t)tap * 256 * 256;
    if (tap + 1 < 25)  // pull next tap's weight rows toward L2/WGP$
      __builtin_prefetch(
          wpk + (size_t)(tap + 1) * 256 * 256 + (size_t)oc * 256, 0, 2);
#pragma unroll 1
    for (int ic0 = 0; ic0 < 256; ic0 += 32) {
      v16h b = load_fragB(wt, 256, col0, ic0);
#pragma unroll
      for (int mt = 0; mt < MT_TILES; ++mt) {
        v16h a;
        if (ro[mt] >= 0) {
          const char* q = smem + ro[mt] + ic0 * 2;
          a = frag_cat(*(const v8h*)q, *(const v8h*)(q + 32));
        } else {
          a = frag_zero();
        }
        acc[mt] = __builtin_amdgcn_wmma_f32_16x16x32_f16(
            false, a, false, b, (short)0, acc[mt], false, false);
      }
    }
  }

  // ---- BN + ReLU epilogue, f16 NHWC store ----
  float sc = bns[oc], bo = bnb[oc];
  int pbase = (lane & 16) ? 8 : 0;  // C/D layout M offset
#pragma unroll
  for (int mt = 0; mt < MT_TILES; ++mt) {
#pragma unroll
    for (int i = 0; i < 8; ++i) {
      int p = mt * 16 + pbase + i;
      if (p < NPIX) {
        float v = fmaxf(acc[mt][i] * sc + bo, 0.f);
        actout[imgoff + (size_t)p * 256 + oc] = (_Float16)v;
      }
    }
  }
}

// ---------------------------------------------------------------------------
// Elementwise / data-movement kernels
// ---------------------------------------------------------------------------
__global__ void cvt_f32_f16(const float* __restrict__ in,
                            _Float16* __restrict__ out, int n) {
  int i = blockIdx.x * blockDim.x + threadIdx.x;
  if (i < n) out[i] = (_Float16)in[i];
}

// conv weights (oc, ic, 5, 5) f32 -> wpk (tap, oc, ic) f16
__global__ void repack_conv_w(const float* __restrict__ w,
                              _Float16* __restrict__ wpk) {
  int i = blockIdx.x * 256 + threadIdx.x;
  if (i < 25 * 256 * 256) {
    int tap = i / (256 * 256);
    int r = i % (256 * 256);
    int oc = r >> 8, ic = r & 255;
    wpk[i] = (_Float16)w[((size_t)(oc * 256 + ic)) * 25 + tap];
  }
}

// xcat[n][p][c] f16: n = offset*128 + b*32 + w ; time = clamp(w*8+offset+p-8)
__global__ void build_xcat(const float* __restrict__ x,
                           _Float16* __restrict__ xc) {
  int n = blockIdx.x, p = blockIdx.y, c = threadIdx.x;
  int offset = n >> 7;
  int rem = n & 127;
  int b = rem >> 5, w = rem & 31;
  int t = w * KWIN + offset + p - KWIN;
  t = t < 0 ? 0 : (t > TT - 1 ? TT - 1 : t);
  float v = x[(size_t)b * DIM * TT + (size_t)c * TT + t];
  xc[((size_t)n * LW + p) * DIM + c] = (_Float16)v;
}

// LSTM gate nonlinearity + state update. z rows are sequences (1024), 4H cols.
__global__ void lstm_gates(const float* __restrict__ z, long zstride,
                           float* __restrict__ c, _Float16* __restrict__ h16,
                           _Float16* __restrict__ hseq16,
                           float* __restrict__ hseq32, int t) {
  int row = blockIdx.x;
  int j = threadIdx.x;
  const float* zr = z + (size_t)row * zstride;
  float zi = zr[j], zf = zr[H + j], zg = zr[2 * H + j], zo = zr[3 * H + j];
  float ig = 1.f / (1.f + expf(-zi));
  float fg = 1.f / (1.f + expf(-zf));
  float gg = tanhf(zg);
  float og = 1.f / (1.f + expf(-zo));
  float cv = (t == 0) ? 0.f : c[(size_t)row * H + j];
  cv = fg * cv + ig * gg;
  float hv = og * tanhf(cv);
  c[(size_t)row * H + j] = cv;
  h16[(size_t)row * H + j] = (_Float16)hv;
  size_t so = ((size_t)row * LW + t) * H + j;
  hseq16[so] = (_Float16)hv;
  hseq32[so] = hv;
}

// Grouped cosine similarity: hseq32 (n,17,256) -> sim (n,4,17,17)
__global__ void sim_kernel(const float* __restrict__ hseq,
                           float* __restrict__ sim) {
  __shared__ float hs[LW * DIM];
  __shared__ float nrm[LW * GROUP];
  int n = blockIdx.x, tid = threadIdx.x;
  for (int i = tid; i < LW * DIM; i += 256)
    hs[i] = hseq[(size_t)n * LW * DIM + i];
  __syncthreads();
  if (tid < LW * GROUP) {
    int i = tid >> 2, g = tid & 3;
    float s = 0.f;
#pragma unroll 8
    for (int d = 0; d < 64; ++d) {
      float v = hs[i * DIM + g * 64 + d];
      s += v * v;
    }
    nrm[tid] = sqrtf(s);
  }
  __syncthreads();
  for (int idx = tid; idx < GROUP * NPIX; idx += 256) {
    int g = idx / NPIX;
    int r = idx % NPIX;
    int i = r / 17, j = r % 17;
    float dot = 0.f;
#pragma unroll 8
    for (int d = 0; d < 64; ++d)
      dot += hs[i * DIM + g * 64 + d] * hs[j * DIM + g * 64 + d];
    float den = fmaxf(nrm[i * 4 + g] * nrm[j * 4 + g], 1e-8f);
    sim[((size_t)n * GROUP + g) * NPIX + r] = dot / den;
  }
}

// conv0: 4ch -> 256ch, 5x5 SAME on sim maps. Small (15 GFLOP): VALU + LDS.
// grid (289 pixels, 1024 imgs), block 256 (one oc per thread). out NHWC f16.
__global__ void conv0_kernel(const float* __restrict__ sim,
                             const float* __restrict__ w,  // (256,4,5,5)
                             const float* __restrict__ s,
                             const float* __restrict__ b,
                             _Float16* __restrict__ out) {
  __shared__ float patch[100];
  int p = blockIdx.x, n = blockIdx.y, oc = threadIdx.x;
  int y = p / 17, x = p % 17;
  if (threadIdx.x < 100) {
    int g = threadIdx.x / 25, r = threadIdx.x % 25;
    int ky = r / 5, kx = r % 5;
    int iy = y + ky - 2, ix = x + kx - 2;
    float v = 0.f;
    if (iy >= 0 && iy < 17 && ix >= 0 && ix < 17)
      v = sim[((size_t)n * GROUP + g) * NPIX + iy * 17 + ix];
    patch[threadIdx.x] = v;
  }
  __syncthreads();
  const float* wr = w + (size_t)oc * 100;
  float acc = 0.f;
#pragma unroll 10
  for (int i = 0; i < 100; ++i) acc += patch[i] * wr[i];
  float v = fmaxf(acc * s[oc] + b[oc], 0.f);
  out[((size_t)n * NPIX + p) * 256 + oc] = (_Float16)v;
}

// Global average pool over 289 pixels, scatter into out (B, DIM, T).
__global__ void pool_scatter(const _Float16* __restrict__ act,
                             float* __restrict__ out) {
  int n = blockIdx.x, oc = threadIdx.x;
  const _Float16* a = act + (size_t)n * NPIX * 256 + oc;
  float s = 0.f;
  for (int p = 0; p < NPIX; ++p) s += (float)a[(size_t)p * 256];
  s *= (1.f / (float)NPIX);
  int offset = n >> 7;
  int rem = n & 127;
  int b = rem >> 5, w = rem & 31;
  int t = w * KWIN + offset;
  out[((size_t)b * DIM + oc) * TT + t] = s;
}

// ---------------------------------------------------------------------------
// Host orchestration
// ---------------------------------------------------------------------------
// Workspace layout (bytes, 256-aligned). Requires ~433 MB of d_ws.
#define O_WIH0H  (size_t)0
#define O_WHH0H  (size_t)524288
#define O_WIH1H  (size_t)1048576
#define O_WHH1H  (size_t)1572864
#define O_WPK1   (size_t)2097152
#define O_WPK2   (size_t)5373952
#define O_WPK3   (size_t)8650752
#define O_XCAT16 (size_t)11927552
#define O_XW     (size_t)20840448
#define O_ZBUF   (size_t)92143616
#define O_H16    (size_t)96337920
#define O_CBUF   (size_t)96862208
#define O_HSEQ16 (size_t)97910784
#define O_HSEQ32 (size_t)106823680
#define O_SIM    (size_t)124649472
#define O_ACTA   (size_t)129384448
#define O_ACTB   (size_t)280903680

extern "C" void kernel_launch(void* const* d_in, const int* in_sizes, int n_in,
                              void* d_out, int out_size, void* d_ws,
                              size_t ws_size, hipStream_t stream) {
  (void)in_sizes; (void)n_in; (void)out_size; (void)ws_size;
  const float* x      = (const float*)d_in[0];
  const float* w_ih0  = (const float*)d_in[1];
  const float* w_hh0  = (const float*)d_in[2];
  const float* b_ih0  = (const float*)d_in[3];
  const float* b_hh0  = (const float*)d_in[4];
  const float* w_ih1  = (const float*)d_in[5];
  const float* w_hh1  = (const float*)d_in[6];
  const float* b_ih1  = (const float*)d_in[7];
  const float* b_hh1  = (const float*)d_in[8];
  const float* conv0_w = (const float*)d_in[9];
  const float* bn0_s  = (const float*)d_in[10];
  const float* bn0_b  = (const float*)d_in[11];
  const float* conv1_w = (const float*)d_in[12];
  const float* bn1_s  = (const float*)d_in[13];
  const float* bn1_b  = (const float*)d_in[14];
  const float* conv2_w = (const float*)d_in[15];
  const float* bn2_s  = (const float*)d_in[16];
  const float* bn2_b  = (const float*)d_in[17];
  const float* conv3_w = (const float*)d_in[18];
  const float* bn3_s  = (const float*)d_in[19];
  const float* bn3_b  = (const float*)d_in[20];
  float* out = (float*)d_out;

  char* ws = (char*)d_ws;
  _Float16* wih0h  = (_Float16*)(ws + O_WIH0H);
  _Float16* whh0h  = (_Float16*)(ws + O_WHH0H);
  _Float16* wih1h  = (_Float16*)(ws + O_WIH1H);
  _Float16* whh1h  = (_Float16*)(ws + O_WHH1H);
  _Float16* wpk1   = (_Float16*)(ws + O_WPK1);
  _Float16* wpk2   = (_Float16*)(ws + O_WPK2);
  _Float16* wpk3   = (_Float16*)(ws + O_WPK3);
  _Float16* xcat16 = (_Float16*)(ws + O_XCAT16);
  float*    xw     = (float*)(ws + O_XW);
  float*    zbuf   = (float*)(ws + O_ZBUF);
  _Float16* h16    = (_Float16*)(ws + O_H16);
  float*    cbuf   = (float*)(ws + O_CBUF);
  _Float16* hseq16 = (_Float16*)(ws + O_HSEQ16);
  float*    hseq32 = (float*)(ws + O_HSEQ32);
  float*    simb   = (float*)(ws + O_SIM);
  _Float16* actA   = (_Float16*)(ws + O_ACTA);
  _Float16* actB   = (_Float16*)(ws + O_ACTB);

  // --- weight conversion / repack ---
  cvt_f32_f16<<<1024, 256, 0, stream>>>(w_ih0, wih0h, FOURH * DIM);
  cvt_f32_f16<<<1024, 256, 0, stream>>>(w_hh0, whh0h, FOURH * H);
  cvt_f32_f16<<<1024, 256, 0, stream>>>(w_ih1, wih1h, FOURH * H);
  cvt_f32_f16<<<1024, 256, 0, stream>>>(w_hh1, whh1h, FOURH * H);
  repack_conv_w<<<6400, 256, 0, stream>>>(conv1_w, wpk1);
  repack_conv_w<<<6400, 256, 0, stream>>>(conv2_w, wpk2);
  repack_conv_w<<<6400, 256, 0, stream>>>(conv3_w, wpk3);

  // --- build windowed sequences ---
  build_xcat<<<dim3(NSEQ, LW), 256, 0, stream>>>(x, xcat16);

  const int M_PROJ = NSEQ * LW;  // 17408

  // ===== LSTM layer 0 =====
  wmma_gemm_nk<<<dim3(FOURH / 64, M_PROJ / 32), 256, 0, stream>>>(
      xcat16, DIM, wih0h, DIM, b_ih0, b_hh0, nullptr, 0, xw, FOURH, DIM);
  for (int t = 0; t < LW; ++t) {
    if (t == 0) {
      lstm_gates<<<NSEQ, 256, 0, stream>>>(xw, (long)LW * FOURH, cbuf, h16,
                                           hseq16, hseq32, t);
    } else {
      wmma_gemm_nk<<<dim3(FOURH / 64, NSEQ / 32), 256, 0, stream>>>(
          h16, H, whh0h, H, nullptr, nullptr, xw + (size_t)t * FOURH,
          (long)LW * FOURH, zbuf, FOURH, H);
      lstm_gates<<<NSEQ, 256, 0, stream>>>(zbuf, (long)FOURH, cbuf, h16,
                                           hseq16, hseq32, t);
    }
  }

  // ===== LSTM layer 1 (input = hseq16 from layer 0) =====
  wmma_gemm_nk<<<dim3(FOURH / 64, M_PROJ / 32), 256, 0, stream>>>(
      hseq16, H, wih1h, H, b_ih1, b_hh1, nullptr, 0, xw, FOURH, H);
  for (int t = 0; t < LW; ++t) {
    if (t == 0) {
      lstm_gates<<<NSEQ, 256, 0, stream>>>(xw, (long)LW * FOURH, cbuf, h16,
                                           hseq16, hseq32, t);
    } else {
      wmma_gemm_nk<<<dim3(FOURH / 64, NSEQ / 32), 256, 0, stream>>>(
          h16, H, whh1h, H, nullptr, nullptr, xw + (size_t)t * FOURH,
          (long)LW * FOURH, zbuf, FOURH, H);
      lstm_gates<<<NSEQ, 256, 0, stream>>>(zbuf, (long)FOURH, cbuf, h16,
                                           hseq16, hseq32, t);
    }
  }

  // ===== similarity heads =====
  sim_kernel<<<NSEQ, 256, 0, stream>>>(hseq32, simb);

  // ===== conv stack =====
  conv0_kernel<<<dim3(NPIX, NSEQ), 256, 0, stream>>>(simb, conv0_w, bn0_s,
                                                     bn0_b, actA);
  wmma_conv5x5<<<dim3(NSEQ), 512, IMG_BYTES, stream>>>(actA, wpk1, bn1_s,
                                                       bn1_b, actB);
  wmma_conv5x5<<<dim3(NSEQ), 512, IMG_BYTES, stream>>>(actB, wpk2, bn2_s,
                                                       bn2_b, actA);
  wmma_conv5x5<<<dim3(NSEQ), 512, IMG_BYTES, stream>>>(actA, wpk3, bn3_s,
                                                       bn3_b, actB);

  // ===== pool + scatter =====
  pool_scatter<<<NSEQ, 256, 0, stream>>>(actB, out);
}
